// Model_37606733644071
// MI455X (gfx1250) — compile-verified, matrix operand order checked by reading, not run on hardware
//
#include <hip/hip_runtime.h>
#include <math.h>

// BiLSTM-CRF for MI455X (gfx1250, wave32, WMMA bf16 -> f32 accum).
// Pipeline: cast weights->bf16 | embed gather->bf16 | pre = x@wih^T+bias (WMMA, whole chip,
//           written in WMMA C/D *fragment-major* layout so the recurrent kernel reads b128)
//           | per-direction recurrent kernel: h in LDS, c in VGPRs, A-frags cached in VGPRs,
//             ~4096 v_wmma/step/WGP, global_prefetch of next step's pre
//           | emissions GEMM (WMMA) | Viterbi: 1 wave per batch row (K=32 == wave32).

#define B_ 128
#define T_ 256
#define E_ 128
#define H_ 256
#define G4 1024   // 4*H
#define K_ 32

typedef __bf16 bf16_t;
typedef __attribute__((ext_vector_type(8)))  float  v8f;
typedef __attribute__((ext_vector_type(16))) __bf16 v16bf;
typedef __attribute__((ext_vector_type(8)))  __bf16 v8bf;

// Build a 16x32 bf16 A/B fragment for v_wmma_f32_16x16x32_bf16.
// Lane L holds row (L&15); K chunks [kb, kb+8) and [kb+16, kb+24) where kb = (L>>4)*8
// relative to the 32-wide K tile. Caller passes pointer at (row, ktile*32 + (L>>4)*8).
static __device__ __forceinline__ v16bf frag16(const bf16_t* p) {
  v8bf lo = *(const v8bf*)(p);
  v8bf hi = *(const v8bf*)(p + 16);
  return __builtin_shufflevector(lo, hi, 0,1,2,3,4,5,6,7,8,9,10,11,12,13,14,15);
}

static __device__ __forceinline__ v8f zero8() {
  v8f z;
#pragma unroll
  for (int i = 0; i < 8; ++i) z[i] = 0.0f;
  return z;
}

static __device__ __forceinline__ float sigmoidf_(float x) {
  return 1.0f / (1.0f + __expf(-x));
}

// ---------------- f32 -> bf16 cast ----------------
__global__ __launch_bounds__(256) void cast_bf16_k(const float* __restrict__ src,
                                                   bf16_t* __restrict__ dst, int n) {
  int i = blockIdx.x * 256 + threadIdx.x;
  if (i < n) dst[i] = (bf16_t)src[i];
}

// ---------------- embedding gather -> bf16 ----------------
__global__ __launch_bounds__(256) void embed_k(const int* __restrict__ tok,
                                               const float* __restrict__ emb,
                                               bf16_t* __restrict__ x) {
  int i = blockIdx.x * 256 + threadIdx.x;          // over B*T*E
  if (i >= B_ * T_ * E_) return;
  int m = i >> 7, e = i & (E_ - 1);
  int t = tok[m];
  x[i] = (bf16_t)emb[(size_t)t * E_ + e];
}

// ---------------- pre = x @ wih^T + bih + bhh  (both directions) ----------------
// M dimension tiled as (t, batch-tile) so tiles line up with the recurrence's batch tiles.
// Output stored fragment-major: pre[d][((t*8 + bt)*64 + ntile)*256 + lane*8 + r] (bf16),
// i.e. each lane's 8 C/D rows are contiguous -> single b128 store here, b128 load in lstm.
__global__ __launch_bounds__(256) void pre_gemm_k(
    const bf16_t* __restrict__ x, const bf16_t* __restrict__ wih2,
    const float* __restrict__ bih_f, const float* __restrict__ bhh_f,
    const float* __restrict__ bih_b, const float* __restrict__ bhh_b,
    bf16_t* __restrict__ pre2) {
  int lane = threadIdx.x & 31;
  int gid  = blockIdx.x * 8 + (threadIdx.x >> 5);  // wave id in [0, 65536)
  int d      = gid >> 15;
  int rem    = gid & 32767;
  int mtile  = rem >> 4;       // 2048 M-tiles: (t, bt)
  int nstrip = rem & 15;       // 16 strips of 64 columns
  int t  = mtile >> 3;
  int bt = mtile & 7;
  int hi = lane >> 4, ln = lane & 15;

  const bf16_t* wih = wih2 + (size_t)d * G4 * E_;
  bf16_t* pre = pre2 + (size_t)d * (size_t)(B_ * T_) * G4;
  const float* bih = d ? bih_b : bih_f;
  const float* bhh = d ? bhh_b : bhh_f;

  v8f acc[4];
#pragma unroll
  for (int s = 0; s < 4; ++s) acc[s] = zero8();

  int rowA = bt * 16 + ln;                         // batch row
#pragma unroll
  for (int kt = 0; kt < 4; ++kt) {
    int kb = kt * 32 + hi * 8;
    v16bf a = frag16(x + ((size_t)rowA * T_ + t) * E_ + kb);
#pragma unroll
    for (int s = 0; s < 4; ++s) {
      int nrow = nstrip * 64 + s * 16 + ln;        // row of wih == column of B
      v16bf b = frag16(wih + (size_t)nrow * E_ + kb);
      acc[s] = __builtin_amdgcn_wmma_f32_16x16x32_bf16(
          false, a, false, b, (short)0, acc[s], false, false);
    }
  }
#pragma unroll
  for (int s = 0; s < 4; ++s) {
    int nt = nstrip * 4 + s;                       // n-tile id 0..63
    int n  = nt * 16 + ln;
    float bias = bih[n] + bhh[n];
    v8bf ov;
#pragma unroll
    for (int r = 0; r < 8; ++r) ov[r] = (bf16_t)(acc[s][r] + bias);
    *(v8bf*)(pre + ((size_t)(t * 8 + bt) * 64 + nt) * 256 + lane * 8) = ov;
  }
}

// ---------------- recurrent LSTM: one workgroup per direction ----------------
// 32 waves; wave w owns batch-tile (w>>2) and 4 h-tiles (w&3)*4+j.  Per step:
//   prefetch next t's pre | cache 8 A-frags of h_{t-1} from LDS in VGPRs (reused 16x)
//   | 128 v_wmma/wave (whh streamed from L2) | gates | barrier | write h | barrier.
__global__ __launch_bounds__(1024) void lstm_rec_k(
    const bf16_t* __restrict__ pre2, const bf16_t* __restrict__ whh2,
    bf16_t* __restrict__ hout2) {
  int d = blockIdx.x;
  const bf16_t* pre = pre2 + (size_t)d * (size_t)(B_ * T_) * G4;
  const bf16_t* whh = whh2 + (size_t)d * G4 * H_;
  bf16_t* hout = hout2 + (size_t)d * (size_t)(B_ * T_) * H_;

  int tid = threadIdx.x;
  int lane = tid & 31, w = tid >> 5;
  int hi = lane >> 4, ln = lane & 15;
  int mt = w >> 2;
  int rowA = mt * 16 + ln;

  __shared__ bf16_t hbuf[B_ * H_];                 // 64 KB: h_{t-1}, bf16 row-major
  for (int i = tid; i < B_ * H_; i += 1024) hbuf[i] = (bf16_t)0.0f;
  __syncthreads();

  float c[4][8];
#pragma unroll
  for (int j = 0; j < 4; ++j)
#pragma unroll
    for (int r = 0; r < 8; ++r) c[j][r] = 0.0f;

  for (int s = 0; s < T_; ++s) {
    int t = d ? (T_ - 1 - s) : s;

    // prefetch next step's pre fragments (global_prefetch_b8) while WMMAs run
    if (s + 1 < T_) {
      int tn = d ? (t - 1) : (t + 1);
#pragma unroll
      for (int j = 0; j < 4; ++j)
#pragma unroll
        for (int g = 0; g < 4; ++g) {
          int nt = g * 16 + ((w & 3) * 4 + j);
          __builtin_prefetch(
              pre + ((size_t)(tn * 8 + mt) * 64 + nt) * 256 + lane * 8, 0, 1);
        }
    }

    // cache all 8 A-fragments of this wave's batch tile (reused by 4 h-tiles x 4 gates)
    v16bf a[8];
#pragma unroll
    for (int kt = 0; kt < 8; ++kt)
      a[kt] = frag16(&hbuf[rowA * H_ + kt * 32 + hi * 8]);

    float hreg[4][8];
#pragma unroll
    for (int j = 0; j < 4; ++j) {
      int ht = (w & 3) * 4 + j;
      v8f acc[4];
      // init accumulators from fragment-major pre: one b128 load per gate
#pragma unroll
      for (int g = 0; g < 4; ++g) {
        v8bf pv = *(const v8bf*)(pre +
            ((size_t)(t * 8 + mt) * 64 + (g * 16 + ht)) * 256 + lane * 8);
#pragma unroll
        for (int r = 0; r < 8; ++r) acc[g][r] = (float)pv[r];
      }
      // h_{t-1} @ whh^T : K = 256 -> 8 k-tiles
#pragma unroll
      for (int kt = 0; kt < 8; ++kt) {
        int kb = kt * 32 + hi * 8;
#pragma unroll
        for (int g = 0; g < 4; ++g) {
          int nrow = g * H_ + ht * 16 + ln;        // row of whh == column of B
          v16bf b = frag16(whh + (size_t)nrow * H_ + kb);
          acc[g] = __builtin_amdgcn_wmma_f32_16x16x32_bf16(
              false, a[kt], false, b, (short)0, acc[g], false, false);
        }
      }
      // gates + cell update (lane-wise; all four gates share the fragment layout)
#pragma unroll
      for (int r = 0; r < 8; ++r) {
        float ig = sigmoidf_(acc[0][r]);
        float fg = sigmoidf_(acc[1][r]);
        float gg = tanhf(acc[2][r]);
        float og = sigmoidf_(acc[3][r]);
        float cn = fg * c[j][r] + ig * gg;
        c[j][r] = cn;
        hreg[j][r] = og * tanhf(cn);
      }
    }

    __syncthreads();                               // all hbuf reads complete
#pragma unroll
    for (int j = 0; j < 4; ++j) {
      int hc = ((w & 3) * 4 + j) * 16 + ln;
#pragma unroll
      for (int r = 0; r < 8; ++r) {
        int bb = mt * 16 + r + 8 * hi;
        bf16_t hv = (bf16_t)hreg[j][r];
        hbuf[bb * H_ + hc] = hv;                             // feed next step
        hout[(size_t)(bb * T_ + t) * H_ + hc] = hv;          // for emissions
      }
    }
    __syncthreads();                               // new h visible next step
  }
}

// ---------------- emissions = [hf, hb] @ w_lin^T + b_lin ----------------
__global__ __launch_bounds__(256) void em_gemm_k(
    const bf16_t* __restrict__ h2, const bf16_t* __restrict__ wlin,
    const float* __restrict__ blin, float* __restrict__ em) {
  int lane = threadIdx.x & 31;
  int mtile = blockIdx.x * 8 + (threadIdx.x >> 5); // 2048 waves
  int hi = lane >> 4, ln = lane & 15;
  const bf16_t* hf = h2;
  const bf16_t* hb = h2 + (size_t)(B_ * T_) * H_;
  int rowA = mtile * 16 + ln;

  v8f acc[2];
#pragma unroll
  for (int s = 0; s < 2; ++s) acc[s] = zero8();

#pragma unroll
  for (int kt = 0; kt < 16; ++kt) {                // K = 512: 0..255 hf, 256..511 hb
    int kb = (kt & 7) * 32 + hi * 8;
    const bf16_t* hsrc = (kt < 8) ? hf : hb;
    v16bf a = frag16(hsrc + (size_t)rowA * H_ + kb);
    int kcol = kt * 32 + hi * 8;
#pragma unroll
    for (int s = 0; s < 2; ++s) {
      int nrow = s * 16 + ln;
      v16bf b = frag16(wlin + (size_t)nrow * 512 + kcol);
      acc[s] = __builtin_amdgcn_wmma_f32_16x16x32_bf16(
          false, a, false, b, (short)0, acc[s], false, false);
    }
  }
#pragma unroll
  for (int s = 0; s < 2; ++s) {
    int n = s * 16 + ln;
    float bias = blin[n];
#pragma unroll
    for (int r = 0; r < 8; ++r) {
      int mo = mtile * 16 + r + 8 * hi;
      em[(size_t)mo * K_ + n] = acc[s][r] + bias;
    }
  }
}

// ---------------- Viterbi: one wave32 per batch row (lane == tag) ----------------
__global__ __launch_bounds__(32) void viterbi_k(
    const float* __restrict__ em, const unsigned char* __restrict__ mask,
    const float* __restrict__ start_t, const float* __restrict__ end_t,
    const float* __restrict__ trans, int* __restrict__ out) {
  int b = blockIdx.x;
  int lane = threadIdx.x;

  __shared__ float trans_s[K_ * K_];
  __shared__ unsigned char hist[(T_ - 1) * K_];    // 8160 B, stays in LDS
  __shared__ float sc[K_];

#pragma unroll 4
  for (int p = 0; p < K_; ++p) trans_s[p * K_ + lane] = trans[p * K_ + lane];

  float score = start_t[lane] + em[(size_t)b * T_ * K_ + lane];
  __syncthreads();

  for (int t = 1; t < T_; ++t) {
    sc[lane] = score;
    __syncthreads();
    float e = em[((size_t)b * T_ + t) * K_ + lane];
    float best = sc[0] + trans_s[lane];
    int arg = 0;
#pragma unroll 4
    for (int p = 1; p < K_; ++p) {
      float cand = sc[p] + trans_s[p * K_ + lane];
      if (cand > best) { best = cand; arg = p; }
    }
    hist[(t - 1) * K_ + lane] = (unsigned char)arg;
    bool m = mask[b * T_ + t] != 0;
    score = m ? (best + e) : score;
    __syncthreads();
  }

  sc[lane] = score + end_t[lane];
  __syncthreads();
  if (lane == 0) {
    float bbest = sc[0]; int cur = 0;
    for (int p = 1; p < K_; ++p) if (sc[p] > bbest) { bbest = sc[p]; cur = p; }
    out[b * T_ + T_ - 1] = mask[b * T_ + T_ - 1] ? cur : 0;
    for (int t = T_ - 2; t >= 0; --t) {
      if (mask[b * T_ + t + 1]) cur = (int)hist[t * K_ + cur];
      out[b * T_ + t] = mask[b * T_ + t] ? cur : 0;
    }
  }
}

extern "C" void kernel_launch(void* const* d_in, const int* in_sizes, int n_in,
                              void* d_out, int out_size, void* d_ws, size_t ws_size,
                              hipStream_t stream) {
  (void)in_sizes; (void)n_in; (void)out_size; (void)ws_size;
  const int*           tok     = (const int*)d_in[0];
  const unsigned char* mask    = (const unsigned char*)d_in[1];
  const float* embed   = (const float*)d_in[2];
  const float* wih_f   = (const float*)d_in[3];
  const float* whh_f   = (const float*)d_in[4];
  const float* bih_f   = (const float*)d_in[5];
  const float* bhh_f   = (const float*)d_in[6];
  const float* wih_b   = (const float*)d_in[7];
  const float* whh_b   = (const float*)d_in[8];
  const float* bih_b   = (const float*)d_in[9];
  const float* bhh_b   = (const float*)d_in[10];
  const float* w_lin   = (const float*)d_in[11];
  const float* b_lin   = (const float*)d_in[12];
  const float* start_t = (const float*)d_in[13];
  const float* end_t   = (const float*)d_in[14];
  const float* trans   = (const float*)d_in[15];
  int* out = (int*)d_out;

  // Workspace layout (~174 MB total)
  char* ws = (char*)d_ws;
  size_t off = 0;
  bf16_t* x_bf    = (bf16_t*)(ws + off); off += (size_t)B_ * T_ * E_ * 2;   // 8 MB
  bf16_t* wih_bf  = (bf16_t*)(ws + off); off += (size_t)2 * G4 * E_ * 2;    // 512 KB
  bf16_t* whh_bf  = (bf16_t*)(ws + off); off += (size_t)2 * G4 * H_ * 2;    // 1 MB
  bf16_t* wlin_bf = (bf16_t*)(ws + off); off += (size_t)K_ * 512 * 2;       // 32 KB
  bf16_t* pre_bf  = (bf16_t*)(ws + off); off += (size_t)2 * B_ * T_ * G4 * 2; // 128 MB
  bf16_t* h_bf    = (bf16_t*)(ws + off); off += (size_t)2 * B_ * T_ * H_ * 2; // 32 MB
  float*  em_f    = (float*)(ws + off);  off += (size_t)B_ * T_ * K_ * 4;   // 4 MB

  // 1) weights -> bf16
  cast_bf16_k<<<(G4 * E_ + 255) / 256, 256, 0, stream>>>(wih_f, wih_bf, G4 * E_);
  cast_bf16_k<<<(G4 * E_ + 255) / 256, 256, 0, stream>>>(wih_b, wih_bf + G4 * E_, G4 * E_);
  cast_bf16_k<<<(G4 * H_ + 255) / 256, 256, 0, stream>>>(whh_f, whh_bf, G4 * H_);
  cast_bf16_k<<<(G4 * H_ + 255) / 256, 256, 0, stream>>>(whh_b, whh_bf + G4 * H_, G4 * H_);
  cast_bf16_k<<<(K_ * 512 + 255) / 256, 256, 0, stream>>>(w_lin, wlin_bf, K_ * 512);

  // 2) embedding gather
  embed_k<<<(B_ * T_ * E_) / 256, 256, 0, stream>>>(tok, embed, x_bf);

  // 3) input projections for both directions (big parallel WMMA GEMM)
  pre_gemm_k<<<8192, 256, 0, stream>>>(x_bf, wih_bf, bih_f, bhh_f, bih_b, bhh_b, pre_bf);

  // 4) sequential recurrence: one workgroup (WGP) per direction
  lstm_rec_k<<<2, 1024, 0, stream>>>(pre_bf, whh_bf, h_bf);

  // 5) emissions
  em_gemm_k<<<256, 256, 0, stream>>>(h_bf, wlin_bf, b_lin, em_f);

  // 6) Viterbi decode
  viterbi_k<<<B_, 32, 0, stream>>>(em_f, mask, start_t, end_t, trans, out);
}
